// MambaLayer_85272280695135
// MI455X (gfx1250) — matrix operand order repaired and checked by reading, not compiled, not run on hardware
//
#include <hip/hip_runtime.h>
#include <hip/hip_bf16.h>

// ---------------- problem constants ----------------
constexpr int kB   = 4;      // batch
constexpr int kL   = 4096;   // sequence length
constexpr int kDIM = 768;    // model dim
constexpr int kDI  = 1536;   // d_inner
constexpr int kE2  = 3072;   // 2*d_inner
constexpr int kRK  = 48;     // dt_rank
constexpr int kG   = 80;     // dt_rank + 2*d_state

// async global->LDS support (gfx1250; guarded so fallback always compiles)
#if defined(__has_builtin)
# if __has_builtin(__builtin_amdgcn_global_load_async_to_lds_b32) && \
     __has_builtin(__builtin_amdgcn_s_wait_asynccnt)
#  define USE_ASYNC_LDS 1
# endif
#endif

// ---------------- types ----------------
typedef __bf16 v16bf __attribute__((ext_vector_type(16)));
typedef float  v8f   __attribute__((ext_vector_type(8)));

static __device__ __forceinline__ unsigned bfbits(float f) {
  unsigned u = __builtin_bit_cast(unsigned, f);
  return ((u + 0x7fffu + ((u >> 16) & 1u)) >> 16) & 0xffffu;  // RNE
}
static __device__ __forceinline__ float dev_silu(float x) {
  return x / (1.0f + __expf(-x));
}
static __device__ __forceinline__ float dev_softplus(float x) {
  return (x > 20.0f) ? x : logf(1.0f + expf(x));
}

// =====================================================================
// Strided GEMM, K-contiguous operands:  C[m,n] = sum_k A[m,k]*B[k,n]
//   A: addr = m*rsA + k        (k stride 1)
//   B: addr = k + n*csB        (k stride 1)
// fp32 -> bf16 at LDS fill; LDS tiles stored in WMMA *fragment layout*
// so each lane reads its v16bf as one contiguous 32B chunk.
// Tile: 64(M) x 128(N), BK=32, 256 threads (8 waves), 4 wmma/wave/step.
// Requires rsA, csB, K even (true at every call site) for float2 loads.
// =====================================================================
__global__ __launch_bounds__(256)
void wmma_gemm_bf16(const float* __restrict__ A, long long rsA, long long bsA,
                    const float* __restrict__ B, long long csB, long long bsB,
                    float*       __restrict__ C, long long rsC, long long csC, long long bsC,
                    int M, int N, int K,
                    const float* __restrict__ bias, int act)
{
  // fragment-layout LDS: [subtile][lane][elem] (elem = v16bf element index)
  __shared__ __attribute__((aligned(32))) __bf16 As[4 * 32 * 16];   // 4 m-subtiles
  __shared__ __attribute__((aligned(32))) __bf16 Bs[8 * 32 * 16];   // 8 n-subtiles

  const int tid  = threadIdx.x;
  const int bz   = blockIdx.z;
  const int m0   = blockIdx.y * 64;
  const int n0   = blockIdx.x * 128;
  A += (long long)bz * bsA;
  B += (long long)bz * bsB;
  C += (long long)bz * bsC;

  const int w    = tid >> 5;       // wave 0..7
  const int lane = tid & 31;
  const int wm   = w & 3;          // m-subtile (16 rows)
  const int wn   = w >> 2;         // n-half (64 cols = 4 n-subtiles)
  const int lm   = lane & 15;
  const int hi   = lane >> 4;

  v8f acc[4] = {};

  for (int k0 = 0; k0 < K; k0 += 32) {
    // ---- stage A (64x32): 1024 float2 pairs, 4/thread, swizzle to frag layout
#pragma unroll
    for (int e = 0; e < 4; ++e) {
      int idx = e * 256 + tid;
      int i = idx >> 4, k = (idx & 15) * 2;
      int gm = m0 + i, gk = k0 + k;
      int gmc = min(gm, M - 1), gkc = min(gk, K - 2);
      const float2 p = *(const float2*)(A + (long long)gmc * rsA + gkc);
      const bool ok = (gm < M) & (gk < K);
      float v0 = ok ? p.x : 0.0f, v1 = ok ? p.y : 0.0f;
      int fhi  = (k >> 3) & 1;                       // which lane-half holds this k
      int j    = ((k & 16) ? 4 : 0) + ((k & 7) >> 1);
      int pos  = (((i >> 4) * 32 + (i & 15) + 16 * fhi) * 16 + 2 * j);
      *(unsigned*)&As[pos] = bfbits(v0) | (bfbits(v1) << 16);
    }
    // ---- stage B (32x128): 2048 pairs, 8/thread
#pragma unroll
    for (int e = 0; e < 8; ++e) {
      int idx = e * 256 + tid;
      int n = idx >> 4, k = (idx & 15) * 2;
      int gn = n0 + n, gk = k0 + k;
      int gnc = min(gn, N - 1), gkc = min(gk, K - 2);
      const float2 p = *(const float2*)(B + (long long)gkc + (long long)gnc * csB);
      const bool ok = (gn < N) & (gk < K);
      float v0 = ok ? p.x : 0.0f, v1 = ok ? p.y : 0.0f;
      int fhi  = (k >> 3) & 1;
      int pos  = (((n >> 4) * 32 + (n & 15) + 16 * fhi) * 16 + ((k & 16) >> 1) + (k & 7));
      *(unsigned*)&Bs[pos] = bfbits(v0) | (bfbits(v1) << 16);
    }
    __syncthreads();

    // speculative prefetch of next K-tile (global_prefetch_b8; bad addr dropped)
    if (k0 + 32 < K) {
      int i = tid >> 2, k = (tid & 3) * 8;
      __builtin_prefetch(A + (long long)min(m0 + i, M - 1) * rsA + (k0 + 32 + k), 0, 1);
      int n = tid >> 1, kb = (tid & 1) * 16;
      __builtin_prefetch(B + (long long)(k0 + 32 + kb) + (long long)min(n0 + n, N - 1) * csB, 0, 1);
    }

    // ---- fragments: single contiguous 32B read per lane
    const v16bf af = *(const v16bf*)&As[(wm * 32 + lane) * 16];
    const v16bf b0 = *(const v16bf*)&Bs[((wn * 4 + 0) * 32 + lane) * 16];
    const v16bf b1 = *(const v16bf*)&Bs[((wn * 4 + 1) * 32 + lane) * 16];
    const v16bf b2 = *(const v16bf*)&Bs[((wn * 4 + 2) * 32 + lane) * 16];
    const v16bf b3 = *(const v16bf*)&Bs[((wn * 4 + 3) * 32 + lane) * 16];

    acc[0] = __builtin_amdgcn_wmma_f32_16x16x32_bf16(false, af, false, b0, (short)0, acc[0], false, false);
    acc[1] = __builtin_amdgcn_wmma_f32_16x16x32_bf16(false, af, false, b1, (short)0, acc[1], false, false);
    acc[2] = __builtin_amdgcn_wmma_f32_16x16x32_bf16(false, af, false, b2, (short)0, acc[2], false, false);
    acc[3] = __builtin_amdgcn_wmma_f32_16x16x32_bf16(false, af, false, b3, (short)0, acc[3], false, false);

    __syncthreads();
  }

  // ---- epilogue: C/D layout n=lane&15, m = 8*(lane>>4) + r
  const int row = m0 + wm * 16 + hi * 8;
#pragma unroll
  for (int r = 0; r < 8; ++r) {
    int rr = row + r;
    if (rr < M) {
      float bv = bias ? bias[rr] : 0.0f;
#pragma unroll
      for (int q = 0; q < 4; ++q) {
        int col = n0 + wn * 64 + q * 16 + lm;
        if (col < N) {
          float v = acc[q][r] + bv;
          if (act) v = dev_softplus(v);
          C[(long long)rr * rsC + (long long)col * csC] = v;
        }
      }
    }
  }
}

// =====================================================================
// silu(z) precompute:  zs[(b*L+l)*DI + d] = silu(xz[(b*L+l)*E2 + DI + d])
// =====================================================================
__global__ __launch_bounds__(256)
void zsilu_kernel(const float* __restrict__ xz, float* __restrict__ zs)
{
  long long i = (long long)blockIdx.x * blockDim.x + threadIdx.x;
  int d = (int)(i % kDI);
  long long bl = i / kDI;
  zs[bl * kDI + d] = dev_silu(xz[bl * kE2 + kDI + d]);
}

// =====================================================================
// causal depthwise conv (k=4) + silu, with optional sequence reversal.
// xz (b,l,e) channel-fastest; xc (b,t,d); map(t) = dir ? L-1-t : t.
// =====================================================================
__global__ __launch_bounds__(256)
void conv_silu_kernel(const float* __restrict__ xz,
                      const float* __restrict__ cw,     // (DI, 1, 4)
                      const float* __restrict__ cb,     // (DI,)
                      float* __restrict__ xc, int dir)
{
  long long i = (long long)blockIdx.x * blockDim.x + threadIdx.x;
  int d = (int)(i % kDI);
  long long r = i / kDI;
  int t = (int)(r % kL);
  int b = (int)(r / kL);

  float acc = cb[d];
#pragma unroll
  for (int j = 0; j < 4; ++j) {
    int s = t - 3 + j;
    if (s >= 0) {
      int l = dir ? (kL - 1 - s) : s;
      acc += cw[d * 4 + j] * xz[((long long)b * kL + l) * kE2 + d];
    }
  }
  xc[((long long)b * kL + t) * kDI + d] = dev_silu(acc);
}

// =====================================================================
// selective scan: 1 lane per (b,d); h[16] in VGPRs; B_t/C_t staged in
// LDS in 16-step chunks via ASYNC global->LDS DMA (ASYNCcnt) when the
// toolchain exposes it.  dir=0 writes comb; dir=1 accumulates reversed.
// =====================================================================
__global__ __launch_bounds__(128)
void scan_kernel(const float* __restrict__ xc,     // u   (b,t,DI)
                 const float* __restrict__ dtb,    // dt  (b,t,DI) post-softplus
                 const float* __restrict__ xdbl,   // (b,t,80): [48:64)=B, [64:80)=C
                 const float* __restrict__ A_log,  // (DI,16)
                 const float* __restrict__ Dp,     // (DI,)
                 const float* __restrict__ zs,     // silu(z) (b,l,DI)
                 float* __restrict__ comb,         // (b,l,DI)
                 int dir)
{
  constexpr int TCH = 16;
  __shared__ float sB[TCH][16];
  __shared__ float sC[TCH][16];

  const int tid = threadIdx.x;
  const int b   = blockIdx.x / (kDI / 128);
  const int d   = (blockIdx.x % (kDI / 128)) * 128 + tid;

  float Arow[16], h[16];
#pragma unroll
  for (int n = 0; n < 16; ++n) {
    Arow[n] = -expf(A_log[d * 16 + n]);
    h[n] = 0.0f;
  }
  const float Dd = Dp[d];
  const long long baseBL = (long long)b * kL;

  for (int t0 = 0; t0 < kL; t0 += TCH) {
#ifdef USE_ASYNC_LDS
    // async DMA: 512 floats (16 steps x 32 chans) global -> LDS, no VGPR hop
#pragma unroll
    for (int e = 0; e < 4; ++e) {
      int idx = e * 128 + tid;
      int tt = idx >> 5, c = idx & 31;
      const float* g = &xdbl[(baseBL + t0 + tt) * kG + 48 + c];
      float* l = (c < 16) ? &sB[tt][c] : &sC[tt][c - 16];
      __builtin_amdgcn_global_load_async_to_lds_b32(
          (__attribute__((address_space(1))) int*)g,
          (__attribute__((address_space(3))) int*)l, 0, 0);
    }
    __builtin_amdgcn_s_wait_asynccnt(0);
    __syncthreads();
#else
#pragma unroll
    for (int e = 0; e < 4; ++e) {
      int idx = e * 128 + tid;
      int tt = idx >> 5, c = idx & 31;
      float v = xdbl[(baseBL + t0 + tt) * kG + 48 + c];
      if (c < 16) sB[tt][c] = v; else sC[tt][c - 16] = v;
    }
    __syncthreads();
#endif

    for (int tt = 0; tt < TCH; ++tt) {
      const int t = t0 + tt;
      const long long off = (baseBL + t) * kDI + d;
      const float u   = xc[off];
      const float dtv = dtb[off];
      float y = 0.0f;
#pragma unroll
      for (int n = 0; n < 16; ++n) {
        float dA = __expf(dtv * Arow[n]);
        h[n] = h[n] * dA + (dtv * sB[tt][n]) * u;
        y = fmaf(h[n], sC[tt][n], y);
      }
      y = fmaf(u, Dd, y);

      const int l = dir ? (kL - 1 - t) : t;
      const long long oidx = (baseBL + l) * kDI + d;
      const float val = y * zs[oidx];
      if (dir) comb[oidx] += val;
      else     comb[oidx]  = val;
    }
    __syncthreads();
  }
}

// =====================================================================
// host-side launch
// =====================================================================
static inline void launch_gemm(const float* A, long long rsA, long long bsA,
                               const float* B, long long csB, long long bsB,
                               float* C, long long rsC, long long csC, long long bsC,
                               int M, int N, int K,
                               const float* bias, int act, hipStream_t stream)
{
  dim3 grid((N + 127) / 128, (M + 63) / 64, kB);
  wmma_gemm_bf16<<<grid, 256, 0, stream>>>(A, rsA, bsA, B, csB, bsB,
                                           C, rsC, csC, bsC, M, N, K, bias, act);
}

extern "C" void kernel_launch(void* const* d_in, const int* in_sizes, int n_in,
                              void* d_out, int out_size, void* d_ws, size_t ws_size,
                              hipStream_t stream)
{
  (void)in_sizes; (void)n_in; (void)out_size;

  const float* x      = (const float*)d_in[0];
  const float* w_in   = (const float*)d_in[1];
  const float* w_out  = (const float*)d_in[2];
  const float* conv_w[2]  = { (const float*)d_in[3],  (const float*)d_in[10] };
  const float* conv_b[2]  = { (const float*)d_in[4],  (const float*)d_in[11] };
  const float* w_xp[2]    = { (const float*)d_in[5],  (const float*)d_in[12] };
  const float* w_dt[2]    = { (const float*)d_in[6],  (const float*)d_in[13] };
  const float* dt_bias[2] = { (const float*)d_in[7],  (const float*)d_in[14] };
  const float* A_log[2]   = { (const float*)d_in[8],  (const float*)d_in[15] };
  const float* Dvec[2]    = { (const float*)d_in[9],  (const float*)d_in[16] };

  // workspace layout (fp32)
  float* ws = (float*)d_ws;
  size_t o = 0;
  float* xz   = ws + o; o += (size_t)kB * kL * kE2;  // (b,l,e)
  float* zs   = ws + o; o += (size_t)kB * kL * kDI;  // silu(z)  (b,l,d)
  float* xc   = ws + o; o += (size_t)kB * kL * kDI;  // conv out (b,t,d)
  float* dtb  = ws + o; o += (size_t)kB * kL * kDI;  // dt       (b,t,d)
  float* xdbl = ws + o; o += (size_t)kB * kL * kG;   // (b,t,80)
  float* comb = ws + o; o += (size_t)kB * kL * kDI;  // out_f + out_r (b,l,d)
  if (ws_size < o * sizeof(float)) return;           // workspace too small: no-op

  const long long LD = (long long)kL * kDIM;
  const long long LE = (long long)kL * kE2;
  const long long LI = (long long)kL * kDI;
  const long long LG = (long long)kL * kG;

  // 1) in_proj: xz[b,l,e] = sum_d in_proj_w[e,d] * x[b,l,d]
  launch_gemm(w_in, kDIM, 0,
              x,    kDIM, LD,
              xz,   1, kE2, LE,
              kE2, kL, kDIM, nullptr, 0, stream);

  // 2) silu(z) precompute
  {
    long long n = (long long)kB * kL * kDI;
    zsilu_kernel<<<(unsigned)(n / 256), 256, 0, stream>>>(xz, zs);
  }

  // 3) both directions (forward writes comb, backward accumulates)
  for (int dir = 0; dir < 2; ++dir) {
    long long n = (long long)kB * kL * kDI;
    conv_silu_kernel<<<(unsigned)(n / 256), 256, 0, stream>>>(xz, conv_w[dir], conv_b[dir], xc, dir);

    // x_dbl[b,t,c] = sum_d x_proj_w[c,d] * xc[b,t,d]
    launch_gemm(w_xp[dir], kDI, 0,
                xc,        kDI, LI,
                xdbl,      1, kG, LG,
                kG, kL, kDI, nullptr, 0, stream);

    // dt[b,t,d] = softplus(sum_r dt_proj_w[d,r] * x_dbl[b,t,r] + dt_bias[d])
    launch_gemm(w_dt[dir], kRK, 0,
                xdbl,      kG, LG,
                dtb,       1, kDI, LI,
                kDI, kL, kRK, dt_bias[dir], 1, stream);

    scan_kernel<<<kB * (kDI / 128), 128, 0, stream>>>(xc, dtb, xdbl, A_log[dir],
                                                      Dvec[dir], zs, comb, dir);
  }

  // 4) out_proj: out[b,l,o] = sum_d comb[b,l,d] * out_proj_w[o,d]
  launch_gemm(comb,  kDI, LI,
              w_out, kDI, 0,
              (float*)d_out, kDIM, 1, LD,
              kL, kDIM, kDI, nullptr, 0, stream);
}
